// RSSM_6562710028805
// MI455X (gfx1250) — matrix operand order repaired and checked by reading, not compiled, not run on hardware
//
#include <hip/hip_runtime.h>
#include <hip/hip_bf16.h>
#include <math.h>
#include <stdint.h>

typedef __bf16 bf16_t;
typedef __attribute__((ext_vector_type(16))) __bf16 v16bf;
typedef __attribute__((ext_vector_type(8)))  __bf16 v8bf;
typedef __attribute__((ext_vector_type(8)))  float  v8f;
typedef int v4i_native __attribute__((vector_size(16)));   // native clang vector (matches builtin)

#define B_SZ   16384
#define H_SZ   512
#define ZF_SZ  1024
#define ACT_SZ 6
#define GIN_SZ 1030
#define KPAD   1056        // GIN padded to multiple of 32
#define N3H    1536

#ifndef __has_builtin
#define __has_builtin(x) 0
#endif

#if __has_builtin(__builtin_amdgcn_global_load_async_to_lds_b128)
#define USE_ASYNC_LDS 1
#else
#define USE_ASYNC_LDS 0
#endif

#if USE_ASYNC_LDS
// builtin signature (from hipcc diagnostic): (v4i AS1*, v4i AS3*, imm int, imm int)
#define AS1P(p) ((__attribute__((address_space(1))) v4i_native*)(p))
#define AS3P(p) ((__attribute__((address_space(3))) v4i_native*)(p))
__device__ __forceinline__ void wait_async0() {
#if __has_builtin(__builtin_amdgcn_s_wait_asynccnt)
    __builtin_amdgcn_s_wait_asynccnt(0);
#else
    asm volatile("s_wait_asynccnt 0x0" ::: "memory");
#endif
}
#endif

// ---------------------------------------------------------------------------
// NT GEMM: C[M,N] = A[M,K] (bf16, row-major) x Bt[N,K]^T (bf16, row-major)
// Block tile 128x128, 8 waves (4x2), wave tile 32x64 = 2x4 WMMA 16x16x32.
// Double-buffered LDS; async-to-LDS staging when available.
// ---------------------------------------------------------------------------
template<bool HAS_BIAS>
__global__ __launch_bounds__(256)
void gemm_nt_bf16(const bf16_t* __restrict__ A, const bf16_t* __restrict__ Bt,
                  const float* __restrict__ bias, float* __restrict__ C,
                  int M, int N, int K)
{
    __shared__ __align__(32) bf16_t lA[2][128 * 32];
    __shared__ __align__(32) bf16_t lB[2][128 * 32];

    const int tid  = threadIdx.x;
    const int lane = tid & 31;
    const int wave = tid >> 5;
    const int wm   = wave >> 1;          // 0..3 (rows of 32)
    const int wn   = wave & 1;           // 0..1 (cols of 64)
    const int bm   = blockIdx.x * 128;
    const int bn   = blockIdx.y * 128;

    v8f acc[2][4] = {};

    // staging: each tile is 128x32 bf16 = 512 16B-chunks; 2 chunks per thread
    const int r0 = tid >> 2;             // 0..63
    const int c0 = (tid & 3) * 8;
    const int lo0 = r0 * 32 + c0;
    const int lo1 = (r0 + 64) * 32 + c0;

    const bf16_t* Ap0 = A  + (size_t)(bm + r0)      * K + c0;
    const bf16_t* Ap1 = A  + (size_t)(bm + r0 + 64) * K + c0;
    const bf16_t* Bp0 = Bt + (size_t)(bn + r0)      * K + c0;
    const bf16_t* Bp1 = Bt + (size_t)(bn + r0 + 64) * K + c0;

    bf16_t* curA = &lA[0][0]; bf16_t* nxtA = &lA[1][0];
    bf16_t* curB = &lB[0][0]; bf16_t* nxtB = &lB[1][0];

    const int kh  = (lane >> 4) * 16;    // A fragment k-phase per lane half
    const int r16 = lane & 15;
    const int bko = (lane >> 4) * 16;    // B fragment k-phase per lane half

    // ---- preload k=0 into buffer 0 ----
#if USE_ASYNC_LDS
    __builtin_amdgcn_global_load_async_to_lds_b128(AS1P(Ap0), AS3P(&curA[lo0]), 0, 0);
    __builtin_amdgcn_global_load_async_to_lds_b128(AS1P(Ap1), AS3P(&curA[lo1]), 0, 0);
    __builtin_amdgcn_global_load_async_to_lds_b128(AS1P(Bp0), AS3P(&curB[lo0]), 0, 0);
    __builtin_amdgcn_global_load_async_to_lds_b128(AS1P(Bp1), AS3P(&curB[lo1]), 0, 0);
    wait_async0();
#else
    *(v4i_native*)&curA[lo0] = *(const v4i_native*)Ap0;
    *(v4i_native*)&curA[lo1] = *(const v4i_native*)Ap1;
    *(v4i_native*)&curB[lo0] = *(const v4i_native*)Bp0;
    *(v4i_native*)&curB[lo1] = *(const v4i_native*)Bp1;
#endif
    __syncthreads();

    for (int k0 = 0; k0 < K; k0 += 32) {
        const bool have_next = (k0 + 32) < K;

#if USE_ASYNC_LDS
        if (have_next) {
            const int kn = k0 + 32;
            __builtin_amdgcn_global_load_async_to_lds_b128(AS1P(Ap0 + kn), AS3P(&nxtA[lo0]), 0, 0);
            __builtin_amdgcn_global_load_async_to_lds_b128(AS1P(Ap1 + kn), AS3P(&nxtA[lo1]), 0, 0);
            __builtin_amdgcn_global_load_async_to_lds_b128(AS1P(Bp0 + kn), AS3P(&nxtB[lo0]), 0, 0);
            __builtin_amdgcn_global_load_async_to_lds_b128(AS1P(Bp1 + kn), AS3P(&nxtB[lo1]), 0, 0);
        }
#else
        v4i_native ra0, ra1, rb0, rb1;
        if (have_next) {
            const int kn = k0 + 32;
            ra0 = *(const v4i_native*)(Ap0 + kn);
            ra1 = *(const v4i_native*)(Ap1 + kn);
            rb0 = *(const v4i_native*)(Bp0 + kn);
            rb1 = *(const v4i_native*)(Bp1 + kn);
            if (k0 + 64 < K) {
                __builtin_prefetch(Ap0 + k0 + 64, 0, 1);
                __builtin_prefetch(Bp0 + k0 + 64, 0, 1);
            }
        }
#endif

        // ---- compute on current buffer ----
        v16bf af[2], bfr[4];
#pragma unroll
        for (int i = 0; i < 2; ++i) {
            // A fragment, contiguous k order (same as B): lane half holds k = kh..kh+15
            const bf16_t* p = &curA[(wm * 32 + i * 16 + r16) * 32 + kh];
            v8bf vlo = *(const v8bf*)p;
            v8bf vhi = *(const v8bf*)(p + 8);
            af[i] = __builtin_shufflevector(vlo, vhi,
                     0, 1, 2, 3, 4, 5, 6, 7, 8, 9, 10, 11, 12, 13, 14, 15);
        }
#pragma unroll
        for (int j = 0; j < 4; ++j) {
            // B fragment, contiguous k order (same as A): lanes 0-15 hold k=0..15, lanes 16-31 k=16..31
            bfr[j] = *(const v16bf*)&curB[(wn * 64 + j * 16 + r16) * 32 + bko];
        }
#pragma unroll
        for (int i = 0; i < 2; ++i)
#pragma unroll
            for (int j = 0; j < 4; ++j)
                acc[i][j] = __builtin_amdgcn_wmma_f32_16x16x32_bf16(
                    false, af[i], false, bfr[j], (short)0, acc[i][j], false, false);

#if USE_ASYNC_LDS
        wait_async0();
#else
        if (have_next) {
            *(v4i_native*)&nxtA[lo0] = ra0;
            *(v4i_native*)&nxtA[lo1] = ra1;
            *(v4i_native*)&nxtB[lo0] = rb0;
            *(v4i_native*)&nxtB[lo1] = rb1;
        }
#endif
        __syncthreads();

        bf16_t* t;
        t = curA; curA = nxtA; nxtA = t;
        t = curB; curB = nxtB; nxtB = t;
    }

    // C/D 16x16 f32 layout: VGPR v -> row v (lanes 0-15) / v+8 (lanes 16-31)
    const int lm = (lane >> 4) * 8;
    const int ln = lane & 15;
#pragma unroll
    for (int i = 0; i < 2; ++i)
#pragma unroll
        for (int j = 0; j < 4; ++j) {
            const int col = bn + wn * 64 + j * 16 + ln;
            const float bv = HAS_BIAS ? bias[col] : 0.0f;
#pragma unroll
            for (int v = 0; v < 8; ++v) {
                const int row = bm + wm * 32 + i * 16 + lm + v;
                C[(size_t)row * N + col] = acc[i][j][v] + bv;
            }
        }
}

// ---------------------------------------------------------------------------
// LayerNorm(ih), LayerNorm(hh), GRU gates -> h; also emit bf16 [h, embed].
// One 256-thread block per batch row.
// ---------------------------------------------------------------------------
__global__ __launch_bounds__(256)
void gru_fuse_kernel(const float* __restrict__ ih, const float* __restrict__ hh,
                     const float* __restrict__ prev_h, const float* __restrict__ embed,
                     const float* __restrict__ g_ih, const float* __restrict__ b_ih,
                     const float* __restrict__ g_hh, const float* __restrict__ b_hh,
                     float* __restrict__ h_out, bf16_t* __restrict__ postin)
{
    __shared__ float sred[256];
    const int row = blockIdx.x, tid = threadIdx.x;
    const float* ihr = ih + (size_t)row * N3H;
    const float* hhr = hh + (size_t)row * N3H;

    float s0 = 0.f, q0 = 0.f, s1 = 0.f, q1 = 0.f;
    for (int j = tid; j < N3H; j += 256) {
        float a = ihr[j]; s0 += a; q0 += a * a;
        float b = hhr[j]; s1 += b; q1 += b * b;
    }
    float vals[4] = { s0, q0, s1, q1 };
    float sums[4];
    for (int t = 0; t < 4; ++t) {
        sred[tid] = vals[t]; __syncthreads();
        for (int s = 128; s > 0; s >>= 1) {
            if (tid < s) sred[tid] += sred[tid + s];
            __syncthreads();
        }
        sums[t] = sred[0]; __syncthreads();
    }
    const float inv = 1.0f / (float)N3H;
    const float m0 = sums[0] * inv, v0 = sums[1] * inv - m0 * m0;
    const float m1 = sums[2] * inv, v1 = sums[3] * inv - m1 * m1;
    const float r0 = rsqrtf(v0 + 1e-5f), r1 = rsqrtf(v1 + 1e-5f);

    for (int j = tid; j < H_SZ; j += 256) {
        float i_r = (ihr[j            ] - m0) * r0 * g_ih[j            ] + b_ih[j            ];
        float i_z = (ihr[j +     H_SZ] - m0) * r0 * g_ih[j +     H_SZ] + b_ih[j +     H_SZ];
        float i_n = (ihr[j + 2 * H_SZ] - m0) * r0 * g_ih[j + 2 * H_SZ] + b_ih[j + 2 * H_SZ];
        float h_r = (hhr[j            ] - m1) * r1 * g_hh[j            ] + b_hh[j            ];
        float h_z = (hhr[j +     H_SZ] - m1) * r1 * g_hh[j +     H_SZ] + b_hh[j +     H_SZ];
        float h_n = (hhr[j + 2 * H_SZ] - m1) * r1 * g_hh[j + 2 * H_SZ] + b_hh[j + 2 * H_SZ];
        float rg = 1.0f / (1.0f + expf(-(i_r + h_r)));
        float zg = 1.0f / (1.0f + expf(-(i_z + h_z)));
        float ng = tanhf(i_n + rg * h_n);
        float ph = prev_h[(size_t)row * H_SZ + j];
        float h  = (1.0f - zg) * ng + zg * ph;
        h_out[(size_t)row * H_SZ + j] = h;
        postin[(size_t)row * (2 * H_SZ) + j]        = (bf16_t)h;
        postin[(size_t)row * (2 * H_SZ) + H_SZ + j] = (bf16_t)embed[(size_t)row * H_SZ + j];
    }
}

// ---------------------------------------------------------------------------
// RMS norm + scale + SiLU -> bf16 u. One block per row. Bias already applied.
// ---------------------------------------------------------------------------
__global__ __launch_bounds__(256)
void rms_silu_kernel(const float* __restrict__ upre, const float* __restrict__ scale,
                     bf16_t* __restrict__ ub)
{
    __shared__ float sred[256];
    const int row = blockIdx.x, tid = threadIdx.x;
    const float* ur = upre + (size_t)row * H_SZ;
    float q = 0.f;
    for (int j = tid; j < H_SZ; j += 256) { float a = ur[j]; q += a * a; }
    sred[tid] = q; __syncthreads();
    for (int s = 128; s > 0; s >>= 1) {
        if (tid < s) sred[tid] += sred[tid + s];
        __syncthreads();
    }
    const float r = rsqrtf(sred[0] / (float)H_SZ + 1e-8f);
    for (int j = tid; j < H_SZ; j += 256) {
        float x = ur[j] * r * scale[j];
        float s = x / (1.0f + expf(-x));
        ub[(size_t)row * H_SZ + j] = (bf16_t)s;
    }
}

// ---------------------------------------------------------------------------
// Prep kernels
// ---------------------------------------------------------------------------
__global__ void pack_gruin_kernel(const float* __restrict__ z, const float* __restrict__ act,
                                  bf16_t* __restrict__ out)
{
    size_t id = (size_t)blockIdx.x * blockDim.x + threadIdx.x;
    if (id >= (size_t)B_SZ * KPAD) return;
    int row = (int)(id / KPAD), col = (int)(id % KPAD);
    float v = 0.f;
    if (col < ZF_SZ)       v = z[(size_t)row * ZF_SZ + col];
    else if (col < GIN_SZ) v = act[(size_t)row * ACT_SZ + (col - ZF_SZ)];
    out[id] = (bf16_t)v;
}

__global__ void cast_f32_bf16_kernel(const float* __restrict__ in, bf16_t* __restrict__ out,
                                     size_t n)
{
    size_t id = (size_t)blockIdx.x * blockDim.x + threadIdx.x;
    if (id < n) out[id] = (bf16_t)in[id];
}

// W[K][N] f32 -> Wt[N][Kpad] bf16, zero-padded tail
__global__ void transpose_pad_kernel(const float* __restrict__ W, bf16_t* __restrict__ Wt,
                                     int K, int N, int Kpad)
{
    size_t id = (size_t)blockIdx.x * blockDim.x + threadIdx.x;
    if (id >= (size_t)N * Kpad) return;
    int n = (int)(id / Kpad), k = (int)(id % Kpad);
    Wt[id] = (k < K) ? (bf16_t)W[(size_t)k * N + n] : (bf16_t)0.f;
}

// ---------------------------------------------------------------------------
extern "C" void kernel_launch(void* const* d_in, const int* in_sizes, int n_in,
                              void* d_out, int out_size, void* d_ws, size_t ws_size,
                              hipStream_t stream)
{
    const float* prev_h  = (const float*)d_in[0];
    const float* prev_z  = (const float*)d_in[1];
    const float* action  = (const float*)d_in[2];
    const float* embed   = (const float*)d_in[3];
    const float* W_ih    = (const float*)d_in[4];
    const float* W_hh    = (const float*)d_in[5];
    const float* ln_ih_g = (const float*)d_in[6];
    const float* ln_ih_b = (const float*)d_in[7];
    const float* ln_hh_g = (const float*)d_in[8];
    const float* ln_hh_b = (const float*)d_in[9];
    const float* mlp_W   = (const float*)d_in[10];
    const float* mlp_b   = (const float*)d_in[11];
    const float* rms_s   = (const float*)d_in[12];
    const float* out_W   = (const float*)d_in[13];
    const float* out_b   = (const float*)d_in[14];

    char*  ws  = (char*)d_ws;
    size_t off = 0;
    auto alloc = [&](size_t bytes) -> char* {
        char* p = ws + off;
        off += (bytes + 255) & ~(size_t)255;
        return p;
    };

    bf16_t* gruin  = (bf16_t*)alloc((size_t)B_SZ * KPAD * 2);    // [B, 1056]
    bf16_t* ph_bf  = (bf16_t*)alloc((size_t)B_SZ * H_SZ * 2);    // [B, 512]
    bf16_t* postin = (bf16_t*)alloc((size_t)B_SZ * 2 * H_SZ * 2);// [B, 1024]
    bf16_t* u_bf   = (bf16_t*)alloc((size_t)B_SZ * H_SZ * 2);    // [B, 512]
    bf16_t* wih_t  = (bf16_t*)alloc((size_t)N3H * KPAD * 2);     // [1536, 1056]
    bf16_t* whh_t  = (bf16_t*)alloc((size_t)N3H * H_SZ * 2);     // [1536, 512]
    bf16_t* mlpw_t = (bf16_t*)alloc((size_t)H_SZ * 2 * H_SZ * 2);// [512, 1024]
    bf16_t* outw_t = (bf16_t*)alloc((size_t)2 * H_SZ * H_SZ * 2);// [1024, 512]
    float*  ih_f   = (float*)alloc((size_t)B_SZ * N3H * 4);      // [B, 1536]
    float*  hh_f   = (float*)alloc((size_t)B_SZ * N3H * 4);      // [B, 1536]
    float*  u_pre  = (float*)ih_f;                               // reuse (ih dead after gates)

    float* h_out  = (float*)d_out;
    float* logits = (float*)d_out + (size_t)B_SZ * H_SZ;

    // --- prep: bf16 packs / weight transposes ---
    {
        size_t n = (size_t)B_SZ * KPAD;
        pack_gruin_kernel<<<dim3((unsigned)((n + 255) / 256)), 256, 0, stream>>>(prev_z, action, gruin);
    }
    {
        size_t n = (size_t)B_SZ * H_SZ;
        cast_f32_bf16_kernel<<<dim3((unsigned)((n + 255) / 256)), 256, 0, stream>>>(prev_h, ph_bf, n);
    }
    {
        size_t n = (size_t)N3H * KPAD;
        transpose_pad_kernel<<<dim3((unsigned)((n + 255) / 256)), 256, 0, stream>>>(W_ih, wih_t, GIN_SZ, N3H, KPAD);
    }
    {
        size_t n = (size_t)N3H * H_SZ;
        transpose_pad_kernel<<<dim3((unsigned)((n + 255) / 256)), 256, 0, stream>>>(W_hh, whh_t, H_SZ, N3H, H_SZ);
    }
    {
        size_t n = (size_t)H_SZ * 2 * H_SZ;
        transpose_pad_kernel<<<dim3((unsigned)((n + 255) / 256)), 256, 0, stream>>>(mlp_W, mlpw_t, 2 * H_SZ, H_SZ, 2 * H_SZ);
        transpose_pad_kernel<<<dim3((unsigned)((n + 255) / 256)), 256, 0, stream>>>(out_W, outw_t, H_SZ, 2 * H_SZ, H_SZ);
    }

    // --- GRU GEMMs: ih = gru_in @ W_ih, hh = prev_h @ W_hh ---
    gemm_nt_bf16<false><<<dim3(B_SZ / 128, N3H / 128), 256, 0, stream>>>(
        gruin, wih_t, nullptr, ih_f, B_SZ, N3H, KPAD);
    gemm_nt_bf16<false><<<dim3(B_SZ / 128, N3H / 128), 256, 0, stream>>>(
        ph_bf, whh_t, nullptr, hh_f, B_SZ, N3H, H_SZ);

    // --- LN + gates -> h (f32 out) and bf16 [h, embed] ---
    gru_fuse_kernel<<<dim3(B_SZ), 256, 0, stream>>>(
        ih_f, hh_f, prev_h, embed, ln_ih_g, ln_ih_b, ln_hh_g, ln_hh_b, h_out, postin);

    // --- posterior MLP: u_pre = post_in @ mlp_W + mlp_b ---
    gemm_nt_bf16<true><<<dim3(B_SZ / 128, H_SZ / 128), 256, 0, stream>>>(
        postin, mlpw_t, mlp_b, u_pre, B_SZ, H_SZ, 2 * H_SZ);

    // --- RMS + silu -> bf16 u ---
    rms_silu_kernel<<<dim3(B_SZ), 256, 0, stream>>>(u_pre, rms_s, u_bf);

    // --- logits = u @ out_W + out_b ---
    gemm_nt_bf16<true><<<dim3(B_SZ / 128, (2 * H_SZ) / 128), 256, 0, stream>>>(
        u_bf, outw_t, out_b, logits, B_SZ, 2 * H_SZ, H_SZ);
}
